// BertAttention_54726473285714
// MI455X (gfx1250) — 16-bit and 8-bit matrix operand order repaired and checked by reading, compiled, not run on hardware
//
#include <hip/hip_runtime.h>
#include <hip/hip_bf16.h>


typedef __attribute__((ext_vector_type(16))) __bf16 v16bf;
typedef __attribute__((ext_vector_type(8)))  float  v8f;
typedef __attribute__((ext_vector_type(8)))  int    v8i;

#define HDIM   1024
#define NHEADS 16
#define HEADD  64
#define BATCH  8
#define SEQ    1024
#define MTOK   (BATCH * SEQ)
#define SQ_ACT (2.5f / 127.0f)
#define INV_SQ (127.0f / 2.5f)

#define OFF_ALPHA 0u
#define OFF_WQKV  256u
#define OFF_WO8   (OFF_WQKV + 6291456u)
#define OFF_Q8    (OFF_WO8 + 1048576u)
#define OFF_K8    (OFF_Q8 + 8388608u)
#define OFF_V8    (OFF_K8 + 8388608u)
#define OFF_CTX8  (OFF_V8 + 8388608u)
#define OFF_XRES  (OFF_CTX8 + 8388608u)

__device__ __forceinline__ void async_copy_b128(unsigned lds_off,
                                                const void* gaddr)
{
    asm volatile("global_load_async_to_lds_b128 %0, %1, off"
                 :
                 : "v"(lds_off), "v"((unsigned long long)(uintptr_t)gaddr)
                 : "memory");
}

__device__ __forceinline__ void wait_asynccnt0()
{
#if __has_builtin(__builtin_amdgcn_s_wait_asynccnt)
    __builtin_amdgcn_s_wait_asynccnt(0);
#else
    asm volatile("s_wait_asynccnt 0x0" ::: "memory");
#endif
}

__global__ __launch_bounds__(256) void absmax_kernel(
    const float* __restrict__ w0, const float* __restrict__ w1,
    const float* __restrict__ w2, const float* __restrict__ w3,
    unsigned* __restrict__ alpha_bits)
{
    const float* wsel = (blockIdx.y == 0) ? w0 : (blockIdx.y == 1) ? w1
                      : (blockIdx.y == 2) ? w2 : w3;
    float m = 0.0f;
    for (int i = blockIdx.x * blockDim.x + threadIdx.x; i < HDIM * HDIM;
         i += gridDim.x * blockDim.x)
        m = fmaxf(m, fabsf(wsel[i]));
    for (int o = 16; o > 0; o >>= 1) m = fmaxf(m, __shfl_xor(m, o, 32));
    __shared__ float red[8];
    const int wave = threadIdx.x >> 5, lane = threadIdx.x & 31;
    if (lane == 0) red[wave] = m;
    __syncthreads();
    if (wave == 0) {
        m = (lane < 8) ? red[lane] : 0.0f;
        for (int o = 16; o > 0; o >>= 1) m = fmaxf(m, __shfl_xor(m, o, 32));
        if (lane == 0) atomicMax(&alpha_bits[blockIdx.y], __float_as_uint(m));
    }
}

__global__ __launch_bounds__(256) void quant_wqkv_kernel(
    const float* __restrict__ Wq, const float* __restrict__ Wk,
    const float* __restrict__ Wv, const unsigned* __restrict__ alpha_bits,
    unsigned short* __restrict__ out)
{
    const int i = blockIdx.x * blockDim.x + threadIdx.x;
    if (i >= 3 * HDIM * HDIM) return;
    const int t = i >> 20;
    const int j = i & (HDIM * HDIM - 1);
    const float* W = (t == 0) ? Wq : (t == 1) ? Wk : Wv;
    const float alpha = __uint_as_float(alpha_bits[t]);
    const float s = alpha * (1.0f / 127.0f);
    const float q = __builtin_rintf(W[j] / s);
    out[i] = (unsigned short)(__float_as_uint(q) >> 16);
}

__global__ __launch_bounds__(256) void quant_wo_kernel(
    const float* __restrict__ Wo, const unsigned* __restrict__ alpha_bits,
    signed char* __restrict__ out)
{
    const int i = blockIdx.x * blockDim.x + threadIdx.x;
    if (i >= HDIM * HDIM) return;
    const float alpha = __uint_as_float(alpha_bits[3]);
    const float s = alpha * (1.0f / 127.0f);
    out[i] = (signed char)(int)__builtin_rintf(Wo[i] / s);
}

__global__ __launch_bounds__(256) void qkv_gemm_kernel(
    const float* __restrict__ x, const unsigned short* __restrict__ wb,
    const float* __restrict__ bq, const float* __restrict__ bk,
    const float* __restrict__ bv, const unsigned* __restrict__ alpha_bits,
    signed char* __restrict__ q8, signed char* __restrict__ k8,
    signed char* __restrict__ v8)
{
    __shared__ unsigned short Xhi[64 * 34];
    __shared__ unsigned short Xlo[64 * 34];
    const int tid   = threadIdx.x;
    const int wave  = tid >> 5, lane = tid & 31;
    const int rtile = wave & 3;
    const int chalf = wave >> 2;
    const int mbase = blockIdx.y * 64;
    const int nbase = blockIdx.x * 128;
    const int t     = nbase >> 10;
    const int lhalf = (lane >= 16) ? 1 : 0;
    const int l15   = lane & 15;

    v8f acc[4];
    #pragma unroll
    for (int c = 0; c < 4; c++)
        #pragma unroll
        for (int r = 0; r < 8; r++) acc[c][r] = 0.0f;

    for (int kb = 0; kb < HDIM; kb += 32) {
        __syncthreads();
        for (int e = tid; e < 64 * 32; e += 256) {
            const int rr = e >> 5, cc = e & 31;
            const float xv = x[(size_t)(mbase + rr) * HDIM + kb + cc];
            unsigned u  = __float_as_uint(xv);
            unsigned hi = (u + 0x7fffu + ((u >> 16) & 1u)) >> 16;
            const float hif = __uint_as_float(hi << 16);
            unsigned ul = __float_as_uint(xv - hif);
            unsigned lo = (ul + 0x7fffu + ((ul >> 16) & 1u)) >> 16;
            Xhi[rr * 34 + cc] = (unsigned short)hi;
            Xlo[rr * 34 + cc] = (unsigned short)lo;
        }
        __syncthreads();

        union { int i[8]; v16bf v; } Ahi, Alo;
        const int mrow = rtile * 16 + l15;
        #pragma unroll
        for (int j = 0; j < 8; j++) {
            const int k0 = (j < 4) ? (j * 2 + lhalf * 8)
                                   : (16 + (j - 4) * 2 + lhalf * 8);
            Ahi.i[j] = *(const int*)&Xhi[mrow * 34 + k0];
            Alo.i[j] = *(const int*)&Xlo[mrow * 34 + k0];
        }
        #pragma unroll
        for (int c = 0; c < 4; c++) {
            union { int i[8]; v16bf v; } Bf;
            const int ng = nbase + chalf * 64 + c * 16 + l15;
            #pragma unroll
            for (int j = 0; j < 8; j++) {
                const int k0 = kb + ((j < 4) ? (j * 2 + lhalf * 8)
                                             : (16 + (j - 4) * 2 + lhalf * 8));
                Bf.i[j] = *(const int*)&wb[(size_t)ng * HDIM + k0];
            }
            if (kb + 32 < HDIM)
                __builtin_prefetch(&wb[(size_t)ng * HDIM + kb + 32], 0, 1);
            acc[c] = __builtin_amdgcn_wmma_f32_16x16x32_bf16(
                false, Alo.v, false, Bf.v, (short)0, acc[c], false, false);
            acc[c] = __builtin_amdgcn_wmma_f32_16x16x32_bf16(
                false, Ahi.v, false, Bf.v, (short)0, acc[c], false, false);
        }
    }

    const float alpha = __uint_as_float(alpha_bits[t]);
    const float sw    = alpha * (1.0f / 127.0f);
    const float* bias = (t == 0) ? bq : (t == 1) ? bk : bv;
    signed char* dst  = (t == 0) ? q8 : (t == 1) ? k8 : v8;
    #pragma unroll
    for (int c = 0; c < 4; c++) {
        const int ng = nbase + chalf * 64 + c * 16 + l15;
        const int nn = ng & (HDIM - 1);
        const int h = nn >> 6, d = nn & 63;
        const float bval = bias[nn];
        #pragma unroll
        for (int r = 0; r < 8; r++) {
            const int row = rtile * 16 + r + lhalf * 8;
            const int mg = mbase + row;
            const int bI = mg >> 10, sI = mg & (SEQ - 1);
            const float y  = acc[c][r] * sw + bval;
            const float xc = fminf(fmaxf(y, -2.5f), 2.5f);
            const int   qi = (int)__builtin_rintf(xc * INV_SQ);
            dst[(size_t)((bI * NHEADS + h) * SEQ + sI) * HEADD + d] =
                (signed char)qi;
        }
    }
}

__global__ __launch_bounds__(256) void attn_kernel(
    const signed char* __restrict__ q8, const signed char* __restrict__ k8,
    const signed char* __restrict__ v8, const float* __restrict__ mask,
    float* __restrict__ scores_out, float* __restrict__ probs_out,
    signed char* __restrict__ ctx8)
{
    __shared__ unsigned char P8[32 * 1024];
    __shared__ signed char   Vt[64 * 64];
    const int tid   = threadIdx.x;
    const int wave  = tid >> 5, lane = tid & 31;
    const int lhalf = (lane >= 16) ? 1 : 0;
    const int l15   = lane & 15;
    const int bh = blockIdx.y, b = bh >> 4, h = bh & 15;
    const int qbase = blockIdx.x * 32;
    const size_t sbase = (size_t)bh << 20;
    const float sscale = SQ_ACT * SQ_ACT * 0.125f;

    {
        const int rtile = wave >> 2;
        const int n0w   = (wave & 3) * 256;
        union { int i[8]; v8i v; } A;
        const int mrow = qbase + rtile * 16 + l15;
        const signed char* qrow = q8 + ((size_t)bh * SEQ + mrow) * HEADD;
        #pragma unroll
        for (int j = 0; j < 8; j++) {
            const int k0 = (j >> 1) * 16 + (j & 1) * 4 + lhalf * 8;
            A.i[j] = *(const int*)(qrow + k0);
        }
        #pragma unroll 4
        for (int c = 0; c < 16; c++) {
            const int n0 = n0w + c * 16;
            union { int i[8]; v8i v; } Bv;
            const signed char* krow =
                k8 + ((size_t)bh * SEQ + (n0 + l15)) * HEADD;
            if (c + 1 < 16)
                __builtin_prefetch(krow + 16 * HEADD, 0, 1);
            #pragma unroll
            for (int j = 0; j < 8; j++) {
                const int k0 = (j >> 1) * 16 + (j & 1) * 4 + lhalf * 8;
                Bv.i[j] = *(const int*)(krow + k0);
            }
            v8i z = {0, 0, 0, 0, 0, 0, 0, 0};
            v8i acci = __builtin_amdgcn_wmma_i32_16x16x64_iu8(
                true, A.v, true, Bv.v, z, false, false);
            const int col = n0 + l15;
            const float mk = mask[b * SEQ + col];
            #pragma unroll
            for (int r = 0; r < 8; r++) {
                const int row = rtile * 16 + r + lhalf * 8;
                const float val = (float)acci[r] * sscale + mk;
                scores_out[sbase + (size_t)(qbase + row) * SEQ + col] = val;
            }
        }
    }
    __threadfence();
    __syncthreads();

    for (int rr = 0; rr < 4; rr++) {
        const int row = wave * 4 + rr;
        const float* srow = scores_out + sbase + (size_t)(qbase + row) * SEQ;
        float mx = -3.4e38f;
        for (int i = lane; i < SEQ; i += 32) mx = fmaxf(mx, srow[i]);
        for (int o = 16; o > 0; o >>= 1) mx = fmaxf(mx, __shfl_xor(mx, o, 32));
        float sum = 0.0f;
        for (int i = lane; i < SEQ; i += 32) sum += __expf(srow[i] - mx);
        for (int o = 16; o > 0; o >>= 1) sum += __shfl_xor(sum, o, 32);
        const float inv = 1.0f / sum;
        float* prow = probs_out + sbase + (size_t)(qbase + row) * SEQ;
        for (int i = lane; i < SEQ; i += 32) {
            const float p = __expf(srow[i] - mx) * inv;
            prow[i] = p;
            const int qi = (int)__builtin_rintf(fminf(p, 2.5f) * INV_SQ);
            P8[row * 1024 + i] = (unsigned char)qi;
        }
    }
    __syncthreads();

    {
        const int rtile = wave >> 2;
        const int d0    = (wave & 3) * 16;
        v8i acci = {0, 0, 0, 0, 0, 0, 0, 0};
        const int mrow = rtile * 16 + l15;
        for (int ks = 0; ks < SEQ; ks += 64) {
            __syncthreads();
            {
                const int vrow = tid >> 2, vcol = (tid & 3) * 16;
                const unsigned lds_off =
                    (unsigned)(uintptr_t)&Vt[vrow * 64 + vcol];
                const signed char* gsrc =
                    v8 + ((size_t)bh * SEQ + ks + vrow) * HEADD + vcol;
                async_copy_b128(lds_off, gsrc);
                wait_asynccnt0();
            }
            __syncthreads();

            union { int i[8]; v8i v; } A;
            #pragma unroll
            for (int j = 0; j < 8; j++) {
                const int k0 = ks + (j >> 1) * 16 + (j & 1) * 4 + lhalf * 8;
                A.i[j] = *(const int*)&P8[mrow * 1024 + k0];
            }
            union { int i[8]; v8i v; } Bv;
            const int n = d0 + l15;
            #pragma unroll
            for (int j = 0; j < 8; j++) {
                const int kk0 = (j >> 1) * 16 + (j & 1) * 4 + lhalf * 8;
                unsigned w0 = 0;
                #pragma unroll
                for (int t4 = 0; t4 < 4; t4++) {
                    const unsigned by =
                        (unsigned char)Vt[(kk0 + t4) * 64 + n];
                    w0 |= by << (8 * t4);
                }
                Bv.i[j] = (int)w0;
            }
            acci = __builtin_amdgcn_wmma_i32_16x16x64_iu8(
                true, A.v, true, Bv.v, acci, false, false);
        }
        const float cscale = SQ_ACT * SQ_ACT;
        #pragma unroll
        for (int r = 0; r < 8; r++) {
            const int row = rtile * 16 + r + lhalf * 8;
            const float cv = (float)acci[r] * cscale;
            const float xc = fminf(fmaxf(cv, -2.5f), 2.5f);
            const int   qi = (int)__builtin_rintf(xc * INV_SQ);
            ctx8[(size_t)(b * SEQ + qbase + row) * HDIM + h * HEADD + d0 + l15]
                = (signed char)qi;
        }
    }
}

__global__ __launch_bounds__(256) void out_gemm_kernel(
    const signed char* __restrict__ ctx8, const signed char* __restrict__ wo8,
    const float* __restrict__ bo, const float* __restrict__ hidden,
    const unsigned* __restrict__ alpha_bits, float* __restrict__ xres)
{
    const int tid   = threadIdx.x;
    const int wave  = tid >> 5, lane = tid & 31;
    const int lhalf = (lane >= 16) ? 1 : 0;
    const int l15   = lane & 15;
    const int rowbase = blockIdx.x * 32;
    const int rtile   = wave >> 2;
    const int cwave   = (wave & 3) * 32;

    v8i acc[16];
    #pragma unroll
    for (int g = 0; g < 16; g++) acc[g] = (v8i){0, 0, 0, 0, 0, 0, 0, 0};

    for (int ks = 0; ks < HDIM; ks += 64) {
        union { int i[8]; v8i v; } A;
        const int m = rowbase + rtile * 16 + l15;
        const signed char* arow = ctx8 + (size_t)m * HDIM + ks;
        #pragma unroll
        for (int j = 0; j < 8; j++) {
            const int k0 = (j >> 1) * 16 + (j & 1) * 4 + lhalf * 8;
            A.i[j] = *(const int*)(arow + k0);
        }
        for (int cg = 0; cg < 8; cg++) {
            #pragma unroll
            for (int tt = 0; tt < 2; tt++) {
                const int n = cg * 128 + cwave + tt * 16 + l15;
                const signed char* brow = wo8 + (size_t)n * HDIM + ks;
                if (ks + 64 < HDIM)
                    __builtin_prefetch(brow + 64, 0, 1);
                union { int i[8]; v8i v; } Bv;
                #pragma unroll
                for (int j = 0; j < 8; j++) {
                    const int k0 = (j >> 1) * 16 + (j & 1) * 4 + lhalf * 8;
                    Bv.i[j] = *(const int*)(brow + k0);
                }
                acc[cg * 2 + tt] = __builtin_amdgcn_wmma_i32_16x16x64_iu8(
                    true, A.v, true, Bv.v, acc[cg * 2 + tt], false, false);
            }
        }
    }

    const float alpha = __uint_as_float(alpha_bits[3]);
    const float scale = SQ_ACT * (alpha * (1.0f / 127.0f));
    #pragma unroll
    for (int g = 0; g < 16; g++) {
        const int col = (g >> 1) * 128 + cwave + (g & 1) * 16 + l15;
        const float bv_ = bo[col];
        #pragma unroll
        for (int r = 0; r < 8; r++) {
            const int row = rtile * 16 + r + lhalf * 8;
            const size_t gi = (size_t)(rowbase + row) * HDIM + col;
            xres[gi] = (float)acc[g][r] * scale + bv_ + hidden[gi];
        }
    }
}

__global__ __launch_bounds__(256) void ln_kernel(
    const float* __restrict__ xres, const float* __restrict__ gamma,
    const float* __restrict__ beta, float* __restrict__ out)
{
    const int row  = blockIdx.x * 8 + (threadIdx.x >> 5);
    const int lane = threadIdx.x & 31;
    const float* xr = xres + (size_t)row * HDIM;
    float s = 0.0f;
    for (int i = lane; i < HDIM; i += 32) s += xr[i];
    for (int o = 16; o > 0; o >>= 1) s += __shfl_xor(s, o, 32);
    const float mu = s * (1.0f / HDIM);
    float v = 0.0f;
    for (int i = lane; i < HDIM; i += 32) { const float d = xr[i] - mu; v += d * d; }
    for (int o = 16; o > 0; o >>= 1) v += __shfl_xor(v, o, 32);
    const float inv = rsqrtf(v * (1.0f / HDIM) + 1e-12f);
    for (int i = lane; i < HDIM; i += 32)
        out[(size_t)row * HDIM + i] = (xr[i] - mu) * inv * gamma[i] + beta[i];
}

extern "C" void kernel_launch(void* const* d_in, const int* in_sizes, int n_in,
                              void* d_out, int out_size, void* d_ws,
                              size_t ws_size, hipStream_t stream)
{
    const float* hidden = (const float*)d_in[0];
    const float* mask   = (const float*)d_in[1];
    const float* Wq = (const float*)d_in[2];  const float* bq = (const float*)d_in[3];
    const float* Wk = (const float*)d_in[4];  const float* bk = (const float*)d_in[5];
    const float* Wv = (const float*)d_in[6];  const float* bv = (const float*)d_in[7];
    const float* Wo = (const float*)d_in[8];  const float* bo = (const float*)d_in[9];
    const float* gamma = (const float*)d_in[10];
    const float* beta  = (const float*)d_in[11];

    float* out        = (float*)d_out;
    float* attn_out   = out;
    float* scores_out = out + 8388608;
    float* probs_out  = out + 8388608 + 134217728;

    char* ws = (char*)d_ws;
    unsigned*       alpha_bits = (unsigned*)(ws + OFF_ALPHA);
    unsigned short* wqkv       = (unsigned short*)(ws + OFF_WQKV);
    signed char*    wo8        = (signed char*)(ws + OFF_WO8);
    signed char*    q8         = (signed char*)(ws + OFF_Q8);
    signed char*    k8         = (signed char*)(ws + OFF_K8);
    signed char*    v8         = (signed char*)(ws + OFF_V8);
    signed char*    ctx8       = (signed char*)(ws + OFF_CTX8);
    float*          xres       = (float*)(ws + OFF_XRES);

    hipMemsetAsync(alpha_bits, 0, 256, stream);

    absmax_kernel<<<dim3(64, 4), 256, 0, stream>>>(Wq, Wk, Wv, Wo, alpha_bits);
    quant_wqkv_kernel<<<(3 * HDIM * HDIM + 255) / 256, 256, 0, stream>>>(
        Wq, Wk, Wv, alpha_bits, wqkv);
    quant_wo_kernel<<<(HDIM * HDIM + 255) / 256, 256, 0, stream>>>(
        Wo, alpha_bits, wo8);
    qkv_gemm_kernel<<<dim3(24, MTOK / 64), 256, 0, stream>>>(
        hidden, wqkv, bq, bk, bv, alpha_bits, q8, k8, v8);
    attn_kernel<<<dim3(SEQ / 32, BATCH * NHEADS), 256, 0, stream>>>(
        q8, k8, v8, mask, scores_out, probs_out, ctx8);
    out_gemm_kernel<<<MTOK / 32, 256, 0, stream>>>(
        ctx8, wo8, bo, hidden, alpha_bits, xres);
    ln_kernel<<<MTOK / 8, 256, 0, stream>>>(xres, gamma, beta, attn_out);
}
